// GraphCapsuleLayer_22832046145641
// MI455X (gfx1250) — matrix operand order repaired and checked by reading, not compiled, not run on hardware
//
#include <hip/hip_runtime.h>
#include <math.h>

// ---------------------------------------------------------------------------
// Fused capsule dynamic-routing kernel for MI455X (gfx1250, wave32).
//
//   u_hat[bn,i,o] = sum_c x[bn,i,c] * W[i,c,o]     v_wmma_f32_16x16x32_bf16
//   3 routing iterations (softmax / contract / squash / b-update) in LDS.
//
// One workgroup = 8 (b,n) positions. u_hat (8*32*256 f32 = 256 KB) stays in
// the 320 KB WGP LDS. W[i] tiles are streamed global->LDS with the Tensor
// Data Mover (tensor_load_to_lds), software-pipelined one tile ahead of the
// WMMA consumer; the output tile is written back with tensor_store_from_lds.
// x and W are re-packed in LDS as bf16 pairs in the exact WMMA A/B VGPR
// layout so fragment loads are raw ds_load_b32/b128 with no per-use convert.
// ---------------------------------------------------------------------------

typedef __attribute__((ext_vector_type(16))) __bf16         v16bf;
typedef __attribute__((ext_vector_type(16))) unsigned short v16us;
typedef __attribute__((ext_vector_type(8)))  float          v8f;
typedef __attribute__((ext_vector_type(4)))  unsigned int   v4u;
typedef __attribute__((ext_vector_type(8)))  int            v8i_t;
typedef __attribute__((ext_vector_type(4)))  int            v4i_t;

union Frag { v16us u16; v16bf bf; unsigned int w[8]; };

#define BN_TOTAL 4096   // B*N = 32*128
#define IC 32           // input capsules
#define ID 16           // input capsule dim
#define NC 16           // output capsules
#define DC 16           // output capsule dim
#define OD 256          // NC*DC
#define P  8            // (b,n) positions per workgroup
#define CAPS_EPS 1e-7f

// LDS layout (float-unit offsets); dynamic LDS starts at byte 0.
#define OFF_UH    0                       // [P][IC][OD] f32      (65536 f)
#define OFF_XPK   (OFF_UH + P*IC*OD)      // [P][IC][8] bf16x2    ( 2048 u32)
#define OFF_BB    (OFF_XPK + P*IC*8)      // [P][NC][IC] f32      ( 4096 f)
#define OFF_CRAW  (OFF_BB + P*NC*IC)      // raw W[i] f32 / c     ( 4096 f)
#define OFF_OO    (OFF_CRAW + ID*OD)      // wpk bf16x2 / outputs ( 2048)
#define LDS_FLOATS (OFF_OO + P*NC*DC)     // 77824 f = 311296 B  (<320 KB)

#if __has_builtin(__builtin_amdgcn_tensor_load_to_lds) && \
    __has_builtin(__builtin_amdgcn_tensor_store_from_lds) && \
    __has_builtin(__builtin_amdgcn_s_wait_tensorcnt)
#define HAVE_TDM 1
#else
#define HAVE_TDM 0
#endif

__device__ __forceinline__ unsigned short f2bf(float f) {
  union { float f; unsigned int u; } c; c.f = f;
  unsigned int r = c.u + 0x7FFFu + ((c.u >> 16) & 1u);   // RNE f32 -> bf16
  return (unsigned short)(r >> 16);
}
__device__ __forceinline__ unsigned int packbf(float lo, float hi) {
  return (unsigned int)f2bf(lo) | ((unsigned int)f2bf(hi) << 16);
}

#if HAVE_TDM
// 1-D contiguous TDM copy of `nelem` 4-byte elements (nelem <= 65535 so the
// whole transfer fits tile_dim0).  D# group0: count=1, lds_addr, global_addr,
// type=2.  group1: data_size=4B, tensor_dim0=tile_dim0=nelem, dims1+ = 1/0.
// This toolchain's builtin is the 6-arg form:
//   (uint32x4 g0, int32x8 g1, int32x4 g2, int32x4 g3, int32x8 g4, i32 cpol)
__device__ __forceinline__ void tdm_copy_1d(bool to_lds, unsigned lds_byte,
                                            const void* gptr, unsigned nelem) {
  unsigned long long ga = (unsigned long long)(uintptr_t)gptr;
  v4u g0;
  g0[0] = 1u;                                              // count=1
  g0[1] = lds_byte;                                        // lds_addr
  g0[2] = (unsigned)ga;                                    // global_addr[31:0]
  g0[3] = ((unsigned)(ga >> 32) & 0x01FFFFFFu) | 0x80000000u; // [56:32] | type=2
  v8i_t g1;
  g1[0] = 0x00020000;                  // workgroup_mask=0, data_size=4B
  g1[1] = (int)(nelem << 16);          // tensor_dim0[15:0]  (bits 63:48)
  g1[2] = (int)((nelem >> 16) | (1u << 16)); // dim0[31:16] | tensor_dim1=1
  g1[3] = (int)(nelem << 16);          // tile_dim0 = nelem  (bits 127:112)
  g1[4] = 1;                           // tile_dim1 = 1, tile_dim2 = 0
  g1[5] = (int)nelem;                  // tensor_dim0_stride[31:0]
  g1[6] = 0;                           // stride hi, tensor_dim1_stride lo
  g1[7] = 0;
  v4i_t z4  = {0, 0, 0, 0};
  v8i_t z8  = {0, 0, 0, 0, 0, 0, 0, 0};
  if (to_lds) __builtin_amdgcn_tensor_load_to_lds(g0, g1, z4, z4, z8, 0);
  else        __builtin_amdgcn_tensor_store_from_lds(g0, g1, z4, z4, z8, 0);
}
#endif

__global__ void capsule_routing_fused(const float* __restrict__ inp,   // [BN, IC, ID]
                                      const float* __restrict__ W,     // [IC, ID, OD]
                                      float* __restrict__ out)         // [BN, NC, DC]
{
  extern __shared__ float smem[];
  float*        uh   = smem + OFF_UH;
  unsigned int* xpk  = (unsigned int*)(smem + OFF_XPK);
  float*        bb   = smem + OFF_BB;
  float*        craw = smem + OFF_CRAW;   // phase1: raw f32 W[i]; phase2: c
  unsigned int* wpk  = (unsigned int*)(smem + OFF_OO); // phase1 packed W[i]
  float*        oo   = smem + OFF_OO;     // phase2 outputs [P][NC][DC]

  const int tid  = threadIdx.x;
  const int lane = tid & 31;
  const int wave = tid >> 5;              // 8 wave32 waves
  const int wg   = blockIdx.x;

#if HAVE_TDM
  // kick TDM for W[0] before anything else (wave 0 owns the TENSORcnt)
  if (wave == 0) tdm_copy_1d(true, OFF_CRAW * 4u, W, ID * OD);
#endif

  // ---- stage x as packed bf16 K-pairs [p][i][vk], zero routing logits -----
  const float* gx = inp + (size_t)wg * (P * IC * ID);
  for (int e = tid; e < P * IC * 8; e += 256)
    xpk[e] = packbf(gx[2 * e], gx[2 * e + 1]);   // K contiguous: pair {2vk,2vk+1}
  for (int e = tid; e < P * NC * IC; e += 256) bb[e] = 0.0f;
  __syncthreads();

  // ---------------- phase 1: u_hat via v_wmma_f32_16x16x32_bf16 ------------
  // A (16x32 bf16): rows = positions (8 valid, 8 zeroed), K=16 padded to 32.
  //   lanes 0-15: VGPR v = K pair {2v,2v+1}; lanes 16-31: {8+2v,8+2v+1}.
  // B (32x16 bf16): lanes 0-15: VGPR v = {K=2v,K=2v+1} at N=lane; lanes 16-31
  //   hold {K=8+2v,K=9+2v}; VGPRs 4-7 (K=16..31) = zero padding.
  for (int i = 0; i < IC; ++i) {
#if HAVE_TDM
    if (wave == 0) __builtin_amdgcn_s_wait_tensorcnt(0);   // W[i] landed
    __syncthreads();
#else
    const float* gw = W + (size_t)i * (ID * OD);
    for (int e = tid; e < ID * OD; e += 256) craw[e] = gw[e];
    __syncthreads();
#endif

    // repack raw f32 W[i] -> bf16 pairs wpk[vk][col] = {K=2vk, K=2vk+1}
    for (int e = tid; e < 8 * OD; e += 256) {
      const int vk = e >> 8, col = e & 255;
      wpk[e] = packbf(craw[vk * 512 + col], craw[vk * 512 + 256 + col]);
    }
    __syncthreads();                                       // wpk ready, craw free

#if HAVE_TDM
    if (wave == 0 && i + 1 < IC)                           // prefetch W[i+1]
      tdm_copy_1d(true, OFF_CRAW * 4u, W + (size_t)(i + 1) * (ID * OD), ID * OD);
#endif

    // A fragment (per i, shared by both column tiles) — one b128 LDS load
    Frag a;
    {
      const int r    = lane & 15;
      const int base = (lane < 16) ? 0 : 4;                // K pairs 0..3 / 4..7
      const unsigned int* xp = xpk + (((r & (P - 1)) * IC + i) << 3);
      const bool valid = (r < P);
      #pragma unroll
      for (int j = 0; j < 4; ++j) a.w[j] = valid ? xp[base + j] : 0u;
      #pragma unroll
      for (int j = 4; j < 8; ++j) a.w[j] = 0u;             // K = 16..31 pad
    }

    #pragma unroll
    for (int tt = 0; tt < 2; ++tt) {
      const int col0 = (wave * 2 + tt) * 16;               // column tile
      Frag b;
      {
        const unsigned int* wp = wpk + col0 + (lane & 15) + (((lane < 16) ? 0 : 4) << 8);
        #pragma unroll
        for (int v = 0; v < 4; ++v) b.w[v] = wp[v << 8];   // VGPR v = K pair base+v
        #pragma unroll
        for (int v = 4; v < 8; ++v) b.w[v] = 0u;           // K = 16..31 pad
      }
      v8f acc = {};
      acc = __builtin_amdgcn_wmma_f32_16x16x32_bf16(
          false, a.bf, false, b.bf, (short)0, acc, false, false);
      if (lane < 16) {                                     // M=v = position v
        #pragma unroll
        for (int v = 0; v < 8; ++v)
          uh[(v * IC + i) * OD + col0 + lane] = acc[v];
      }
    }
    __syncthreads();
  }

  // ---------------- phase 2: dynamic routing (all in LDS) ------------------
  float* cc = craw;                     // reuse raw-W buffer as c [P][NC][IC]
  for (int it = 0; it < 3; ++it) {
    // c = softmax(b, axis=k): one (p,i) per thread (8*32 = 256)
    {
      const int p  = tid >> 5;
      const int i0 = tid & 31;
      float vals[NC];
      float mx = -1e30f;
      #pragma unroll
      for (int k = 0; k < NC; ++k) {
        const float v = bb[(p * NC + k) * IC + i0];
        vals[k] = v; mx = fmaxf(mx, v);
      }
      float s = 0.0f;
      #pragma unroll
      for (int k = 0; k < NC; ++k) { vals[k] = __expf(vals[k] - mx); s += vals[k]; }
      const float inv = 1.0f / s;
      #pragma unroll
      for (int k = 0; k < NC; ++k) cc[(p * NC + k) * IC + i0] = vals[k] * inv;
    }
    __syncthreads();

    // outputs[p,k,d] = sum_i c[p,k,i] * u_hat[p,i,k*16+d]
    for (int e = tid; e < P * NC * DC; e += 256) {
      const int p = e >> 8, k = (e >> 4) & 15, d = e & 15;
      const float* uhp = uh + (p * IC) * OD + k * DC + d;
      const float* cp  = cc + (p * NC + k) * IC;
      float s = 0.0f;
      #pragma unroll
      for (int i = 0; i < IC; ++i) s += cp[i] * uhp[i * OD];
      oo[e] = s;
    }
    __syncthreads();

    if (it < 2) {
      // squash along d: one (p,k) per thread
      if (tid < P * NC) {
        const int p = tid >> 4, k = tid & 15;
        float* op = oo + (p * NC + k) * DC;
        float s2 = 0.0f;
        #pragma unroll
        for (int d = 0; d < DC; ++d) s2 += op[d] * op[d];
        const float sc = s2 / ((1.0f + s2) * sqrtf(s2 + CAPS_EPS));
        #pragma unroll
        for (int d = 0; d < DC; ++d) op[d] *= sc;
      }
      __syncthreads();
      // b[p,k,i] += sum_d outputs[p,k,d] * u_hat[p,i,k*16+d]
      for (int e = tid; e < P * NC * IC; e += 256) {
        const int p = e >> 9, k = (e >> 5) & 15, i = e & 31;
        const float* uhp = uh + (p * IC + i) * OD + k * DC;
        const float* op  = oo + (p * NC + k) * DC;
        float s = 0.0f;
        #pragma unroll
        for (int d = 0; d < DC; ++d) s += op[d] * uhp[d];
        bb[e] += s;
      }
      __syncthreads();
    }
  }

  // ---------------- write final (un-squashed) outputs ----------------------
#if HAVE_TDM
  if (wave == 0) {
    tdm_copy_1d(false, OFF_OO * 4u, out + (size_t)wg * (P * NC * DC), P * NC * DC);
    __builtin_amdgcn_s_wait_tensorcnt(0);
  }
#else
  float* go = out + (size_t)wg * (P * NC * DC);
  for (int e = tid; e < P * NC * DC; e += 256) go[e] = oo[e];
#endif
}

extern "C" void kernel_launch(void* const* d_in, const int* in_sizes, int n_in,
                              void* d_out, int out_size, void* d_ws, size_t ws_size,
                              hipStream_t stream) {
  (void)in_sizes; (void)n_in; (void)d_ws; (void)ws_size; (void)out_size;
  const float* inp = (const float*)d_in[0];   // [32,128,32,16] fp32
  const float* W   = (const float*)d_in[1];   // [32,16,256]    fp32
  float*       out = (float*)d_out;           // [32,128,16,16] fp32

  const size_t shmem = (size_t)LDS_FLOATS * sizeof(float);  // 311296 B

  dim3 grid(BN_TOTAL / P);   // 512 workgroups
  dim3 block(256);           // 8 wave32 waves
  hipLaunchKernelGGL(capsule_routing_fused, grid, block, shmem, stream,
                     inp, W, out);
}
